// PL_CNN_2_43319040147639
// MI455X (gfx1250) — matrix operand order repaired and checked by reading, not compiled, not run on hardware
//
#include <hip/hip_runtime.h>
#include <math.h>

typedef __attribute__((ext_vector_type(16))) _Float16 v16h;
typedef __attribute__((ext_vector_type(8)))  float    v8f;

#define HGRID 28
#define NPIX 784
#define BATCH 32
#define TPTS 32

// ---------------------------------------------------------------------------
// 1. Build _NN_IDX / _D2_SORTED on device (stable argsort via rank counting).
//    d2 values are small integers -> exact fp32; ties broken by index, which
//    matches the stable sort used by np.argsort / XLA.
// ---------------------------------------------------------------------------
__global__ void build_nn_kernel(int* __restrict__ nn, float* __restrict__ d2s) {
  int p = blockIdx.x;                // 0..783
  int ip = p / HGRID, jp = p % HGRID;
  for (int q = threadIdx.x; q < NPIX; q += blockDim.x) {
    int iq = q / HGRID, jq = q % HGRID;
    int d2 = (ip - iq) * (ip - iq) + (jp - jq) * (jp - jq);
    int rank = 0;
    for (int r = 0; r < NPIX; ++r) {
      int ir = r / HGRID, jr = r % HGRID;
      int dr = (ip - ir) * (ip - ir) + (jp - jr) * (jp - jr);
      rank += (dr < d2) || (dr == d2 && r < q);
    }
    nn[p * NPIX + rank]  = q;
    d2s[p * NPIX + rank] = (float)d2;
  }
}

// ---------------------------------------------------------------------------
// 2. conv1: [B,1,28,28] -> relu -> [B,32,28,28]
// ---------------------------------------------------------------------------
__global__ void conv1_kernel(const float* __restrict__ x, const float* __restrict__ w,
                             const float* __restrict__ bias, float* __restrict__ y) {
  int tid = blockIdx.x * blockDim.x + threadIdx.x;
  if (tid >= BATCH * 32 * NPIX) return;
  int p  = tid % NPIX;
  int oc = (tid / NPIX) % 32;
  int b  = tid / (NPIX * 32);
  int h = p / HGRID, wj = p % HGRID;
  float acc = bias[oc];
  #pragma unroll
  for (int dh = 0; dh < 3; ++dh)
    #pragma unroll
    for (int dw = 0; dw < 3; ++dw) {
      int hh = h + dh - 1, ww = wj + dw - 1;
      if (hh >= 0 && hh < HGRID && ww >= 0 && ww < HGRID)
        acc += x[b * NPIX + hh * HGRID + ww] * w[oc * 9 + dh * 3 + dw];
    }
  y[(b * 32 + oc) * NPIX + p] = fmaxf(acc, 0.f);
}

// conv2: [B,32,28,28] -> [B,1,28,28]; relu (part of concat relu) into xcat cols 0..783
__global__ void conv2_kernel(const float* __restrict__ x1, const float* __restrict__ w,
                             const float* __restrict__ bias, float* __restrict__ xcat) {
  int tid = blockIdx.x * blockDim.x + threadIdx.x;
  if (tid >= BATCH * NPIX) return;
  int p = tid % NPIX, b = tid / NPIX;
  int h = p / HGRID, wj = p % HGRID;
  float acc = bias[0];
  for (int ic = 0; ic < 32; ++ic) {
    const float* xi = x1 + (b * 32 + ic) * NPIX;
    const float* wk = w + ic * 9;
    #pragma unroll
    for (int dh = 0; dh < 3; ++dh)
      #pragma unroll
      for (int dw = 0; dw < 3; ++dw) {
        int hh = h + dh - 1, ww = wj + dw - 1;
        if (hh >= 0 && hh < HGRID && ww >= 0 && ww < HGRID)
          acc += xi[hh * HGRID + ww] * wk[dh * 3 + dw];
      }
  }
  xcat[b * 848 + p] = fmaxf(acc, 0.f);
}

// ---------------------------------------------------------------------------
// 3. per-image pixel-weight sums, then DTM
// ---------------------------------------------------------------------------
__global__ void rowsum_kernel(const float* __restrict__ x, float* __restrict__ sums) {
  __shared__ float s[256];
  int b = blockIdx.x;
  float acc = 0.f;
  for (int p = threadIdx.x; p < NPIX; p += 256) acc += x[b * NPIX + p];
  s[threadIdx.x] = acc;
  __syncthreads();
  for (int st = 128; st > 0; st >>= 1) {
    if ((int)threadIdx.x < st) s[threadIdx.x] += s[threadIdx.x + st];
    __syncthreads();
  }
  if (threadIdx.x == 0) sums[b] = s[0];
}

__global__ void dtm_kernel(const float* __restrict__ x, const int* __restrict__ nn,
                           const float* __restrict__ d2s, const float* __restrict__ sums,
                           float m0, float* __restrict__ out) {
  int tid = blockIdx.x * blockDim.x + threadIdx.x;
  if (tid >= BATCH * NPIX) return;
  int b = tid / NPIX, p = tid % NPIX;
  float bound = m0 * sums[b];
  const int*   nnp = nn  + p * NPIX;
  const float* dp  = d2s + p * NPIX;
  const float* xb  = x + b * NPIX;
  float cw = 0.f, cwd = 0.f, val = 0.f;
  bool found = false;
  for (int j = 0; j < NPIX; ++j) {
    float w = xb[nnp[j]];
    float ncw = cw + w;
    if (ncw >= bound) {            // k = j (first index where cumsum >= bound)
      val = (cwd + (bound - cw) * dp[j]) / bound;   // cw/cwd hold cumsum[k-1]
      found = true;
      break;
    }
    cwd += w * dp[j];
    cw = ncw;
  }
  if (!found) val = 0.f;           // argmax over all-False -> k=0 -> d2k=0 -> val=0
  out[tid] = sqrtf(fmaxf(val, 0.f));
}

// ---------------------------------------------------------------------------
// 4. PH0 union-find. 128 runs = b(32) x field{d1,d2} x sub{+v/NB4, -v/NB8}.
//    Rank (stable argsort) computed in parallel; union-find serial on lane 0.
// ---------------------------------------------------------------------------
__device__ __forceinline__ int uf_find(int* par, int x) {
  while (par[x] != x) { par[x] = par[par[x]]; x = par[x]; }
  return x;
}

__global__ void ph0_kernel(const float* __restrict__ d1, const float* __restrict__ d2,
                           int* __restrict__ pb, int* __restrict__ pd) {
  __shared__ float v[NPIX];
  __shared__ int pos[NPIX];
  __shared__ int order[NPIX];
  __shared__ int parent[NPIX];
  int run   = blockIdx.x;
  int b     = run >> 2;
  int field = (run >> 1) & 1;
  int s     = run & 1;
  const float* src = (field ? d2 : d1) + b * NPIX;
  float sign = s ? -1.f : 1.f;
  for (int i = threadIdx.x; i < NPIX; i += blockDim.x) {
    v[i] = sign * src[i];
    parent[i] = -1;
  }
  __syncthreads();
  for (int i = threadIdx.x; i < NPIX; i += blockDim.x) {
    float vi = v[i];
    int r = 0;
    for (int j = 0; j < NPIX; ++j) {
      float vj = v[j];
      r += (vj < vi) || (vj == vi && j < i);
    }
    pos[i] = r;
    order[r] = i;
  }
  __syncthreads();
  if (threadIdx.x == 0) {
    const int dioff[8] = {-1, 1, 0, 0, -1, -1, 1, 1};
    const int djoff[8] = { 0, 0,-1, 1, -1,  1,-1, 1};
    int Kn = s ? 8 : 4;
    int* PB = pb + run * 6272;
    int* PD = pd + run * 6272;
    for (int step = 0; step < NPIX; ++step) {
      int p = order[step];
      parent[p] = p;
      int pi = p / HGRID, pj = p % HGRID;
      for (int kk = 0; kk < Kn; ++kk) {
        int bi = -1, di = -1;
        int ni = pi + dioff[kk], nj = pj + djoff[kk];
        if (ni >= 0 && ni < HGRID && nj >= 0 && nj < HGRID) {
          int q = ni * HGRID + nj;
          if (parent[q] != -1) {
            int r  = uf_find(parent, q);
            int rp = uf_find(parent, p);
            if (r != rp) {
              int elder   = (pos[r] < pos[rp]) ? r  : rp;
              int younger = (pos[r] < pos[rp]) ? rp : r;
              parent[younger] = elder;
              bi = younger;
              di = p;
            }
          }
        }
        PB[step * Kn + kk] = bi;
        PD[step * Kn + kk] = di;
      }
    }
  }
}

// ---------------------------------------------------------------------------
// 5. Landscapes: per (b, field, feat) block, 32 threads = 32 t samples,
//    each keeps top-3 of the tent values over all pairs.
// ---------------------------------------------------------------------------
__global__ void land_kernel(const float* __restrict__ d1, const float* __restrict__ d2,
                            const int* __restrict__ pb, const int* __restrict__ pd,
                            float* __restrict__ f1, float* __restrict__ f2) {
  int blk   = blockIdx.x;
  int b     = blk >> 2;
  int field = (blk >> 1) & 1;
  int feat  = blk & 1;                  // 0: (b0,d0) from +v/NB4 ; 1: (d1r,b1r) from -v/NB8
  int t     = threadIdx.x;              // 0..31
  const float* v = (field ? d2 : d1) + b * NPIX;
  float start = field ? 1.f : 0.f;
  float end   = field ? 8.f : 7.f;
  int   Kmax  = field ? 3 : 2;
  float ts = start + (end - start) * ((float)t / 31.f);
  int Kn = feat ? 8 : 4;
  int npairs = NPIX * Kn;
  const int* biA = feat ? (pd + blk * 6272) : (pb + blk * 6272);  // feat1 swaps (death,birth)
  const int* diA = feat ? (pb + blk * 6272) : (pd + blk * 6272);
  float t0 = 0.f, t1 = 0.f, t2 = 0.f;   // tents are >= 0; >Kmax exact zeros exist
  for (int e = 0; e < npairs; ++e) {
    int bi = biA[e], di = diA[e];
    if (bi >= 0 && di >= 0) {
      float tent = fminf(ts - v[bi], v[di] - ts);
      tent = fmaxf(tent, 0.f);
      if (tent > t0)      { t2 = t1; t1 = t0; t0 = tent; }
      else if (tent > t1) { t2 = t1; t1 = tent; }
      else if (tent > t2) { t2 = tent; }
    }
  }
  float* outp = field ? (f2 + b * 192 + feat * 96) : (f1 + b * 128 + feat * 64);
  outp[0 * TPTS + t] = t0;
  outp[1 * TPTS + t] = t1;
  if (Kmax == 3) outp[2 * TPTS + t] = t2;
}

// ---------------------------------------------------------------------------
// 6a. Fragment packing: convert f32 matrices into f16 buffers laid out in
//     the exact WMMA operand order [ktile][tile][lane][16 halves], zero-
//     padded at edges so the MMA loop needs no guards.
// ---------------------------------------------------------------------------
// A (16x32 f16 fragment): lanes 0-15/16-31 -> M rows 0-15; per lane,
// halves 0..7 -> K {0,1,2,3,4,5,6,7} (+8 for hi lanes), halves 8..15 ->
// K {16..23} (+8 for hi lanes).
__global__ void pack_a_kernel(const float* __restrict__ A, int lda, int M, int K,
                              _Float16* __restrict__ out, int mtiles, int ktiles) {
  int tid = blockIdx.x * blockDim.x + threadIdx.x;
  int total = mtiles * ktiles * 512;
  if (tid >= total) return;
  int h    = tid & 15;
  int lane = (tid >> 4) & 31;
  int tile = tid >> 9;                  // kt*mtiles + mt
  int mt = tile % mtiles;
  int kt = tile / mtiles;
  int hi = lane >> 4, l15 = lane & 15;
  int vv = h >> 1, lo = h & 1;
  int m = mt * 16 + l15;
  int k = kt * 32 + (vv < 4 ? 2 * vv + lo : 16 + 2 * (vv - 4) + lo) + (hi ? 8 : 0);
  float val = (m < M && k < K) ? A[m * lda + k] : 0.f;
  out[tid] = (_Float16)val;
}

// B = W^T (32x16 f16 fragment): lane (mod 16) = N column; halves 0..7 ->
// K 0..7 (+8 for hi lanes), halves 8..15 -> K 16..23 (+8 for hi lanes).
__global__ void pack_b_kernel(const float* __restrict__ W, int ldw, int N, int K,
                              _Float16* __restrict__ out, int ntiles, int ktiles) {
  int tid = blockIdx.x * blockDim.x + threadIdx.x;
  int total = ntiles * ktiles * 512;
  if (tid >= total) return;
  int h    = tid & 15;
  int lane = (tid >> 4) & 31;
  int tile = tid >> 9;                  // kt*ntiles + nt
  int nt = tile % ntiles;
  int kt = tile / ntiles;
  int hi = lane >> 4, l15 = lane & 15;
  int n = nt * 16 + l15;
  int k = kt * 32 + (h < 8 ? h : 16 + (h - 8)) + hi * 8;
  float val = (n < N && k < K) ? W[n * ldw + k] : 0.f;
  out[tid] = (_Float16)val;
}

// ---------------------------------------------------------------------------
// 6b. WMMA GEMM on packed fragments: C[M,N] = A @ W^T + bias (optional relu).
//     One wave per 16x16 C tile; per K-step: two 32B contiguous loads + one
//     v_wmma_f32_16x16x32_f16. No divergence before the MMA (EXEC all 1s).
// ---------------------------------------------------------------------------
__global__ __launch_bounds__(32)
void wmma_gemm_packed(const _Float16* __restrict__ pa, const _Float16* __restrict__ pbm,
                      const float* __restrict__ bias, float* __restrict__ C, int ldc,
                      int M, int N, int ktiles, int mtiles, int ntiles, int relu) {
  const int lane = threadIdx.x & 31;
  const int tn = blockIdx.x;
  const int tm = blockIdx.y;
  const v16h* PA = (const v16h*)pa;
  const v16h* PB = (const v16h*)pbm;
  v8f c = {};
  for (int kt = 0; kt < ktiles; ++kt) {
    v16h a = PA[(kt * mtiles + tm) * 32 + lane];
    v16h b = PB[(kt * ntiles + tn) * 32 + lane];
    c = __builtin_amdgcn_wmma_f32_16x16x32_f16(false, a, false, b,
                                               (short)0, c, false, false);
  }
  const int hi = lane >> 4, l15 = lane & 15;
  #pragma unroll
  for (int r = 0; r < 8; ++r) {
    int m = tm * 16 + hi * 8 + r;       // C: lanes 0-15 M 0-7, lanes 16-31 M 8-15
    int n = tn * 16 + l15;
    if (m < M && n < N) {
      float val = c[r] + bias[n];
      if (relu) val = fmaxf(val, 0.f);
      C[m * ldc + n] = val;
    }
  }
}

// ---------------------------------------------------------------------------
extern "C" void kernel_launch(void* const* d_in, const int* in_sizes, int n_in,
                              void* d_out, int out_size, void* d_ws, size_t ws_size,
                              hipStream_t stream) {
  (void)in_sizes; (void)n_in; (void)out_size; (void)ws_size;
  const float* x   = (const float*)d_in[0];   // [32,1,28,28]
  const float* w1  = (const float*)d_in[1];   // [32,1,3,3]
  const float* b1  = (const float*)d_in[2];   // [32]
  const float* w2  = (const float*)d_in[3];   // [1,32,3,3]
  const float* b2  = (const float*)d_in[4];   // [1]
  const float* g1w = (const float*)d_in[5];   // [32,128]
  const float* g1b = (const float*)d_in[6];   // [32]
  const float* g2w = (const float*)d_in[7];   // [32,192]
  const float* g2b = (const float*)d_in[8];   // [32]
  const float* f1w = (const float*)d_in[9];   // [64,848]
  const float* f1b = (const float*)d_in[10];  // [64]
  const float* f2w = (const float*)d_in[11];  // [10,64]
  const float* f2b = (const float*)d_in[12];  // [10]
  float* out = (float*)d_out;                 // [32,10]

  char* wsb = (char*)d_ws;
  size_t off = 0;
  auto alloc = [&](size_t bytes) -> char* {
    char* p = wsb + off;
    off = (off + bytes + 255) & ~(size_t)255;
    return p;
  };
  int*   nn   = (int*)  alloc((size_t)NPIX * NPIX * 4);
  float* d2s  = (float*)alloc((size_t)NPIX * NPIX * 4);
  float* x1   = (float*)alloc((size_t)BATCH * 32 * NPIX * 4);
  float* sums = (float*)alloc(BATCH * 4);
  float* d1   = (float*)alloc(BATCH * NPIX * 4);
  float* d2b  = (float*)alloc(BATCH * NPIX * 4);
  int*   pb   = (int*)  alloc((size_t)128 * 6272 * 4);
  int*   pd   = (int*)  alloc((size_t)128 * 6272 * 4);
  float* f1   = (float*)alloc(BATCH * 128 * 4);
  float* f2   = (float*)alloc(BATCH * 192 * 4);
  float* xcat = (float*)alloc(BATCH * 848 * 4);
  float* h1   = (float*)alloc(BATCH * 64 * 4);
  // packed fragment buffers (f16), tiles: mtiles=2 everywhere (M=32)
  _Float16* paG1 = (_Float16*)alloc((size_t)2 * 4  * 512 * 2);  // 32x128
  _Float16* pbG1 = (_Float16*)alloc((size_t)2 * 4  * 512 * 2);
  _Float16* paG2 = (_Float16*)alloc((size_t)2 * 6  * 512 * 2);  // 32x192
  _Float16* pbG2 = (_Float16*)alloc((size_t)2 * 6  * 512 * 2);
  _Float16* paF1 = (_Float16*)alloc((size_t)2 * 27 * 512 * 2);  // 32x848 (pad to 864)
  _Float16* pbF1 = (_Float16*)alloc((size_t)4 * 27 * 512 * 2);  // 64x848
  _Float16* paF2 = (_Float16*)alloc((size_t)2 * 2  * 512 * 2);  // 32x64
  _Float16* pbF2 = (_Float16*)alloc((size_t)1 * 2  * 512 * 2);  // 10x64 (pad N to 16)

  // constant tables
  build_nn_kernel<<<NPIX, 256, 0, stream>>>(nn, d2s);
  // conv path
  conv1_kernel<<<(BATCH * 32 * NPIX + 255) / 256, 256, 0, stream>>>(x, w1, b1, x1);
  conv2_kernel<<<(BATCH * NPIX + 255) / 256, 256, 0, stream>>>(x1, w2, b2, xcat);
  // DTM path
  rowsum_kernel<<<BATCH, 256, 0, stream>>>(x, sums);
  dtm_kernel<<<(BATCH * NPIX + 255) / 256, 256, 0, stream>>>(x, nn, d2s, sums, 0.05f, d1);
  dtm_kernel<<<(BATCH * NPIX + 255) / 256, 256, 0, stream>>>(x, nn, d2s, sums, 0.20f, d2b);
  // persistence + landscapes
  ph0_kernel<<<128, 256, 0, stream>>>(d1, d2b, pb, pd);
  land_kernel<<<128, 32, 0, stream>>>(d1, d2b, pb, pd, f1, f2);

  // ---- dense tail: pack + WMMA ----
  // g1: x2 = relu(f1 @ g1w^T + g1b) -> xcat cols 784..815
  pack_a_kernel<<<(2 * 4 * 512 + 255) / 256, 256, 0, stream>>>(f1, 128, 32, 128, paG1, 2, 4);
  pack_b_kernel<<<(2 * 4 * 512 + 255) / 256, 256, 0, stream>>>(g1w, 128, 32, 128, pbG1, 2, 4);
  wmma_gemm_packed<<<dim3(2, 2), 32, 0, stream>>>(paG1, pbG1, g1b, xcat + 784, 848,
                                                  32, 32, 4, 2, 2, 1);
  // g2: x3 = relu(f2 @ g2w^T + g2b) -> xcat cols 816..847
  pack_a_kernel<<<(2 * 6 * 512 + 255) / 256, 256, 0, stream>>>(f2, 192, 32, 192, paG2, 2, 6);
  pack_b_kernel<<<(2 * 6 * 512 + 255) / 256, 256, 0, stream>>>(g2w, 192, 32, 192, pbG2, 2, 6);
  wmma_gemm_packed<<<dim3(2, 2), 32, 0, stream>>>(paG2, pbG2, g2b, xcat + 816, 848,
                                                  32, 32, 6, 2, 2, 1);
  // fc1: h1 = relu(xcat @ f1w^T + f1b)   (K=848 zero-padded to 864)
  pack_a_kernel<<<(2 * 27 * 512 + 255) / 256, 256, 0, stream>>>(xcat, 848, 32, 848, paF1, 2, 27);
  pack_b_kernel<<<(4 * 27 * 512 + 255) / 256, 256, 0, stream>>>(f1w, 848, 64, 848, pbF1, 4, 27);
  wmma_gemm_packed<<<dim3(4, 2), 32, 0, stream>>>(paF1, pbF1, f1b, h1, 64,
                                                  32, 64, 27, 2, 4, 1);
  // fc2: out = h1 @ f2w^T + f2b          (N=10 zero-padded to 16)
  pack_a_kernel<<<(2 * 2 * 512 + 255) / 256, 256, 0, stream>>>(h1, 64, 32, 64, paF2, 2, 2);
  pack_b_kernel<<<(1 * 2 * 512 + 255) / 256, 256, 0, stream>>>(f2w, 64, 10, 64, pbF2, 1, 2);
  wmma_gemm_packed<<<dim3(1, 2), 32, 0, stream>>>(paF2, pbF2, f2b, out, 10,
                                                  32, 10, 2, 2, 1, 0);
}
